// SELA_18382460027443
// MI455X (gfx1250) — matrix operand order repaired and checked by reading, not compiled, not run on hardware
//
#include <hip/hip_runtime.h>
#include <hip/hip_bf16.h>
#include <stdint.h>

typedef _Float16 f16;
typedef __attribute__((ext_vector_type(16))) _Float16 v16h;
typedef __attribute__((ext_vector_type(8)))  _Float16 v8h;
typedef __attribute__((ext_vector_type(8)))  float    v8f;

#define CCH 256          // channels
#define LL  4096         // h*w
#define NB  8            // batch
#define GRPS 16

// ---- CDNA5 async global->LDS path (guarded; falls back to sync copies) ----
#if defined(__has_builtin)
#if __has_builtin(__builtin_amdgcn_global_load_async_to_lds_b128)
#define HAVE_ASYNC_LDS 1
#endif
#endif

#ifndef HAVE_ASYNC_LDS
#define HAVE_ASYNC_LDS 0
#endif

#if HAVE_ASYNC_LDS
// builtin signature (from hipcc diagnostic): param0 = AS(1) int __vector(4)*,
// non-const; param1 = AS(3) pointer.
typedef int gvec4i __attribute__((__vector_size__(16)));
typedef __attribute__((address_space(1))) gvec4i* glb_v4p;
typedef __attribute__((address_space(3))) gvec4i* lds_v4p;
__device__ inline lds_v4p to_lds(void* p) {
  // generic->LDS: low 32 bits of a generic LDS address are the LDS offset
  return (lds_v4p)(uint32_t)(uintptr_t)p;
}
__device__ inline glb_v4p to_glb(const void* p) {
  return (glb_v4p)(uintptr_t)p;
}
#if __has_builtin(__builtin_amdgcn_s_wait_asynccnt)
#define WAIT_ASYNC(n) __builtin_amdgcn_s_wait_asynccnt(n)
#else
#define WAIT_ASYNC(n) asm volatile("s_wait_asynccnt %0" ::"i"(n) : "memory")
#endif
#endif

__device__ inline v8f zero8() {
  v8f r;
#pragma unroll
  for (int i = 0; i < 8; i++) r[i] = 0.0f;
  return r;
}

// A fragment (16-bit, 16x32, row M = lane%16): two contiguous 8-half runs at
// offsets 8*half and 16+8*half of the row (K contiguous in memory).
__device__ inline v16h load_a16(const f16* rowPtr, int half) {
  v8h lo = *(const v8h*)(rowPtr + 8 * half);
  v8h hi = *(const v8h*)(rowPtr + 16 + 8 * half);
  v16h r;
#pragma unroll
  for (int i = 0; i < 8; i++) { r[i] = lo[i]; r[8 + i] = hi[i]; }
  return r;
}

// B fragment (16-bit, 32x16, col N = lane%16) from a [N][K]-major source:
// per lane, column N fixed, two contiguous 8-half runs at K = 8*half and 16+8*half.
__device__ inline v16h load_b16(const f16* colPtr, int half) {
  v8h lo = *(const v8h*)(colPtr + 8 * half);
  v8h hi = *(const v8h*)(colPtr + 16 + 8 * half);
  v16h r;
#pragma unroll
  for (int i = 0; i < 8; i++) { r[i] = lo[i]; r[8 + i] = hi[i]; }
  return r;
}

__device__ inline v8f wmma16(v16h a, v16h b, v8f c) {
  return __builtin_amdgcn_wmma_f32_16x16x32_f16(false, a, false, b, (short)0, c,
                                                false, false);
}

// ---------------------------------------------------------------- kernel 1
__global__ void k_means(const float* __restrict__ x, float* __restrict__ xh,
                        float* __restrict__ xw) {
  const int bc = blockIdx.x;          // 0..2047
  const int t  = threadIdx.x;         // 0..63
  const float* p = x + (size_t)bc * 4096;
  float rs = 0.f, cs = 0.f;
  for (int j = 0; j < 64; j++) { rs += p[t * 64 + j]; cs += p[j * 64 + t]; }
  xh[(size_t)bc * 64 + t] = rs * (1.f / 64.f);
  xw[(size_t)bc * 64 + t] = cs * (1.f / 64.f);
}

// ---------------------------------------------------------------- kernel 2
__global__ void k_wcast(const float* __restrict__ w, f16* __restrict__ w16) {
  int i = blockIdx.x * 256 + threadIdx.x;      // grid sized exactly
  w16[i] = (f16)w[i];
}

// ---------------------------------------------------------------- kernel 3
// ELA branch: z = w1 @ y + b1 (256x256 * 256x64), GroupNorm(16), sigmoid.
__global__ __launch_bounds__(256) void k_ela(
    const float* __restrict__ xh_in, const float* __restrict__ xw_in,
    const float* __restrict__ w1, const float* __restrict__ b1,
    const float* __restrict__ gamma, const float* __restrict__ beta,
    float* __restrict__ x_h, float* __restrict__ x_w) {
  __shared__ float y_s[256][64];
  __shared__ float gsum[256], gsum2[256];
  __shared__ float gmu[GRPS], grs[GRPS];
  const int b  = blockIdx.x >> 1;
  const int br = blockIdx.x & 1;
  const float* yb = (br ? xw_in : xh_in) + (size_t)b * 256 * 64;
  float*       ob = (br ? x_w   : x_h)   + (size_t)b * 256 * 64;
  const int d = threadIdx.x;

  for (int j = 0; j < 64; j += 4)
    *(float4*)&y_s[d][j] = *(const float4*)(yb + d * 64 + j);
  __syncthreads();

  float z[64];
  const float bias = b1[d];
#pragma unroll
  for (int l = 0; l < 64; l++) z[l] = bias;
  const float* wrow = w1 + (size_t)d * 256;
  for (int c = 0; c < 256; c++) {
    const float wv = wrow[c];
#pragma unroll
    for (int l = 0; l < 64; l++) z[l] += wv * y_s[c][l];
  }
  float s = 0.f, s2 = 0.f;
#pragma unroll
  for (int l = 0; l < 64; l++) { s += z[l]; s2 += z[l] * z[l]; }
  gsum[d] = s; gsum2[d] = s2;
  __syncthreads();
  if (d < GRPS) {
    float a = 0.f, a2 = 0.f;
    for (int j = 0; j < 16; j++) { a += gsum[d * 16 + j]; a2 += gsum2[d * 16 + j]; }
    const float mu = a * (1.f / 1024.f);
    gmu[d] = mu;
    grs[d] = rsqrtf(a2 * (1.f / 1024.f) - mu * mu + 1e-5f);
  }
  __syncthreads();
  const float mu = gmu[d >> 4], rs = grs[d >> 4], g = gamma[d], be = beta[d];
#pragma unroll
  for (int l = 0; l < 64; l++) {
    const float zn = (z[l] - mu) * rs;
    ob[(size_t)d * 64 + l] = 1.f / (1.f + __expf(-(zn * g + be)));
  }
}

// ---------------------------------------------------------------- kernel 4
// QKV projection with WMMA.  Computes (W x)^T tiles: A = x^T (LDS-transposed),
// B = W rows ([N=d][K=c]-major -> contiguous).  B fragments are register
// double-buffered so the 8 WMMAs of K-step kk overlap the loads of kk+1.
// Writes qT[n][c] (scaled), kT[m][c], v[c][m] in f16.
__global__ __launch_bounds__(256) void k_qkv(
    const float* __restrict__ x, const f16* __restrict__ w16,
    f16* __restrict__ qT, f16* __restrict__ kT, f16* __restrict__ vv) {
  __shared__ f16 xT_s[128][CCH + 8];
  const int b = blockIdx.y, nt = blockIdx.x;
  const int tid = threadIdx.x, wv = tid >> 5, lane = tid & 31;
  const int half = lane >> 4, l16 = lane & 15;
  const int n0 = nt * 128;

  // stage x tile transposed + cast to f16: thread = channel row c
  {
    const float* xr = x + ((size_t)b * CCH + tid) * LL + n0;
    for (int j = 0; j < 128; j += 4) {
      float4 v4 = *(const float4*)(xr + j);
      xT_s[j + 0][tid] = (f16)v4.x; xT_s[j + 1][tid] = (f16)v4.y;
      xT_s[j + 2][tid] = (f16)v4.z; xT_s[j + 3][tid] = (f16)v4.w;
    }
  }
  __syncthreads();

  const int nw = n0 + wv * 16;           // wave's n-row base
  v16h xa[8];
#pragma unroll
  for (int kk = 0; kk < 8; kk++)
    xa[kk] = load_a16(&xT_s[wv * 16 + l16][32 * kk], half);

  f16* qTb = qT + (size_t)b * LL * CCH;
  f16* kTb = kT + (size_t)b * LL * CCH;
  f16* vb  = vv + (size_t)b * CCH * LL;
  const float scale = 0.0625f;           // 256^-0.5 folded into q

  for (int dc = 0; dc < 6; dc++) {       // 6 chunks of 128 output channels
    const f16* wbase = w16 + (size_t)(dc * 128 + l16) * CCH;
    v8f acc[8];
#pragma unroll
    for (int ct = 0; ct < 8; ct++) acc[ct] = zero8();

    // prologue: B frags of K-step 0
    v16h bw[8];
#pragma unroll
    for (int ct = 0; ct < 8; ct++)
      bw[ct] = load_b16(wbase + (size_t)(ct * 16) * CCH, half);

#pragma unroll
    for (int kk = 0; kk < 8; kk++) {
      v16h bn[8];
      if (kk < 7) {
#pragma unroll
        for (int ct = 0; ct < 8; ct++)
          bn[ct] = load_b16(wbase + (size_t)(ct * 16) * CCH + 32 * (kk + 1), half);
      }
#pragma unroll
      for (int ct = 0; ct < 8; ct++)
        acc[ct] = wmma16(xa[kk], bw[ct], acc[ct]);
      if (kk < 7) {
#pragma unroll
        for (int ct = 0; ct < 8; ct++) bw[ct] = bn[ct];
      }
    }

    // epilogue: C-frag element r -> (M = n = nw + r + 8*half, N = d = l16-col)
    if (dc < 2) {
#pragma unroll
      for (int ct = 0; ct < 8; ct++) {
        const int d = dc * 128 + ct * 16 + l16;
#pragma unroll
        for (int r = 0; r < 8; r++)
          qTb[(size_t)(nw + r + 8 * half) * CCH + d] = (f16)(acc[ct][r] * scale);
      }
    } else if (dc < 4) {
#pragma unroll
      for (int ct = 0; ct < 8; ct++) {
        const int d = (dc - 2) * 128 + ct * 16 + l16;
#pragma unroll
        for (int r = 0; r < 8; r++)
          kTb[(size_t)(nw + r + 8 * half) * CCH + d] = (f16)acc[ct][r];
      }
    } else {
#pragma unroll
      for (int ct = 0; ct < 8; ct++) {
        const int cch = (dc - 4) * 128 + ct * 16 + l16;
        v8h pk;
#pragma unroll
        for (int r = 0; r < 8; r++) pk[r] = (f16)acc[ct][r];
        *(v8h*)(vb + (size_t)cch * LL + nw + 8 * half) = pk;   // 8 contiguous m
      }
    }
  }
}

// ---------------------------------------------------------------- kernel 5
// Flash attention, double-buffered async global->LDS staging.
// 8 waves/WG; wave owns 16 n-rows, full c=256 accumulator (16 v8f frags).
__global__ __launch_bounds__(256) void k_flash(
    const f16* __restrict__ qT, const f16* __restrict__ kT,
    const f16* __restrict__ vv, float* __restrict__ attn_out) {
  __shared__ f16 kT_s[2][32][CCH + 8];  // [m][c]  ([N][K] for S-matmul B)
  __shared__ f16 v_s[2][CCH][32 + 8];   // [c][m]  ([N][K] for O-matmul B)
  __shared__ f16 p_s[8][16][32 + 8];    // per-wave P tile, row-major [n][m]

  const int b = blockIdx.y, nt = blockIdx.x;
  const int tid = threadIdx.x, wv = tid >> 5, lane = tid & 31;
  const int half = lane >> 4, l16 = lane & 15;
  const int n0 = nt * 128 + wv * 16;

  const f16* qTb = qT + (size_t)b * LL * CCH;
  const f16* kTb = kT + (size_t)b * LL * CCH;
  const f16* vb  = vv + (size_t)b * CCH * LL;

  // per-thread staging slice: 4 b128 of kT + 4 b128 of v per tile
  const int mrel = tid >> 3, part = tid & 7;

  auto stage = [&](int buf, int m0) {
    const f16* ksrc = kTb + (size_t)(m0 + mrel) * CCH + part * 32;
    f16*       kdst = &kT_s[buf][mrel][part * 32];
    const f16* vsrc = vb + (size_t)tid * LL + m0;
    f16*       vdst = &v_s[buf][tid][0];
#if HAVE_ASYNC_LDS
#pragma unroll
    for (int j = 0; j < 4; j++)
      __builtin_amdgcn_global_load_async_to_lds_b128(to_glb(ksrc + j * 8),
                                                     to_lds(kdst + j * 8), 0, 0);
#pragma unroll
    for (int j = 0; j < 4; j++)
      __builtin_amdgcn_global_load_async_to_lds_b128(to_glb(vsrc + j * 8),
                                                     to_lds(vdst + j * 8), 0, 0);
#else
#pragma unroll
    for (int j = 0; j < 4; j++) *(v8h*)(kdst + j * 8) = *(const v8h*)(ksrc + j * 8);
#pragma unroll
    for (int j = 0; j < 4; j++) *(v8h*)(vdst + j * 8) = *(const v8h*)(vsrc + j * 8);
#endif
  };

  v16h qa[8];
#pragma unroll
  for (int kk = 0; kk < 8; kk++)
    qa[kk] = load_a16(qTb + (size_t)(n0 + l16) * CCH + 32 * kk, half);

  v8f o[16];
#pragma unroll
  for (int ct = 0; ct < 16; ct++) o[ct] = zero8();
  float mrow[8], lrow[8];
#pragma unroll
  for (int r = 0; r < 8; r++) { mrow[r] = -1e30f; lrow[r] = 0.f; }

  constexpr int T = LL / 32;            // 128 m-tiles
  stage(0, 0);                          // prologue: tile 0 -> buf 0

  for (int it = 0; it < T; it++) {
    const int cur = it & 1;
    if (it + 1 < T) {
      stage(cur ^ 1, (it + 1) * 32);    // prefetch next tile into other buffer
#if HAVE_ASYNC_LDS
      WAIT_ASYNC(8);                    // retire current tile's 8 loads (in-order)
#endif
    } else {
#if HAVE_ASYNC_LDS
      WAIT_ASYNC(0);
#endif
    }
    __syncthreads();                    // all waves' staging of `cur` visible

    // S = qT x k : two 16x16 tiles (m columns 0-15 and 16-31)
    v8f s0 = zero8(), s1 = zero8();
#pragma unroll
    for (int kk = 0; kk < 8; kk++) {
      s0 = wmma16(qa[kk], load_b16(&kT_s[cur][l16][32 * kk], half), s0);
      s1 = wmma16(qa[kk], load_b16(&kT_s[cur][16 + l16][32 * kk], half), s1);
    }

    // online softmax (rows of a C-frag live within a 16-lane half)
#pragma unroll
    for (int r = 0; r < 8; r++) {
      float mx = fmaxf(s0[r], s1[r]);
#pragma unroll
      for (int dm = 1; dm < 16; dm <<= 1) mx = fmaxf(mx, __shfl_xor(mx, dm, 32));
      const float mnew = fmaxf(mrow[r], mx);
      const float corr = __expf(mrow[r] - mnew);
      const float p0 = __expf(s0[r] - mnew);
      const float p1 = __expf(s1[r] - mnew);
      float ps = p0 + p1;
#pragma unroll
      for (int dm = 1; dm < 16; dm <<= 1) ps += __shfl_xor(ps, dm, 32);
      lrow[r] = lrow[r] * corr + ps;
      mrow[r] = mnew;
#pragma unroll
      for (int ct = 0; ct < 16; ct++) o[ct][r] *= corr;
      const int Mr = r + 8 * half;
      p_s[wv][Mr][l16]      = (f16)p0;
      p_s[wv][Mr][16 + l16] = (f16)p1;
    }
    asm volatile("s_wait_dscnt 0" ::: "memory");  // wave-private P visible

    const v16h pa = load_a16(&p_s[wv][l16][0], half);
#pragma unroll
    for (int ct = 0; ct < 16; ct++)
      o[ct] = wmma16(pa, load_b16(&v_s[cur][16 * ct + l16][0], half), o[ct]);

    __syncthreads();                    // done reading `cur` before re-stage
  }

  // normalize + store attn_out[b][c][n] (8 contiguous n per lane -> 2 float4)
  float inv[8];
#pragma unroll
  for (int r = 0; r < 8; r++) inv[r] = 1.f / lrow[r];
  float* outb = attn_out + (size_t)b * CCH * LL;
#pragma unroll
  for (int ct = 0; ct < 16; ct++) {
    const int c = 16 * ct + l16;
    float* p = outb + (size_t)c * LL + n0 + 8 * half;
    *(float4*)p       = make_float4(o[ct][0] * inv[0], o[ct][1] * inv[1],
                                    o[ct][2] * inv[2], o[ct][3] * inv[3]);
    *(float4*)(p + 4) = make_float4(o[ct][4] * inv[4], o[ct][5] * inv[5],
                                    o[ct][6] * inv[6], o[ct][7] * inv[7]);
  }
}

// ---------------------------------------------------------------- kernel 6
__global__ void k_final(const float* __restrict__ x, const float* __restrict__ x_h,
                        const float* __restrict__ x_w, const float* __restrict__ attn,
                        float* __restrict__ out) {
  const size_t i = (size_t)blockIdx.x * 256 + threadIdx.x;
  const int wq = (int)(i & 63);
  const size_t t = i >> 6;
  const int hq = (int)(t & 63);
  const size_t bc = t >> 6;
  out[i] = x[i] * x_h[bc * 64 + hq] * x_w[bc * 64 + wq] + attn[i];
}

extern "C" void kernel_launch(void* const* d_in, const int* in_sizes, int n_in,
                              void* d_out, int out_size, void* d_ws, size_t ws_size,
                              hipStream_t stream) {
  (void)in_sizes; (void)n_in; (void)out_size; (void)ws_size;
  const float* x     = (const float*)d_in[0];
  const float* w_qkv = (const float*)d_in[1];
  const float* w1    = (const float*)d_in[2];
  const float* b1    = (const float*)d_in[3];
  const float* gam   = (const float*)d_in[4];
  const float* bet   = (const float*)d_in[5];
  float* out = (float*)d_out;

  char* ws = (char*)d_ws;
  size_t off = 0;
  auto alloc = [&](size_t bytes) -> void* {
    off = (off + 255) & ~(size_t)255;
    void* p = ws + off;
    off += bytes;
    return p;
  };
  float* xh_in = (float*)alloc((size_t)NB * CCH * 64 * 4);
  float* xw_in = (float*)alloc((size_t)NB * CCH * 64 * 4);
  float* x_h   = (float*)alloc((size_t)NB * CCH * 64 * 4);
  float* x_w   = (float*)alloc((size_t)NB * CCH * 64 * 4);
  f16*   w16   = (f16*)  alloc((size_t)768 * CCH * 2);
  f16*   qT    = (f16*)  alloc((size_t)NB * LL * CCH * 2);
  f16*   kT    = (f16*)  alloc((size_t)NB * LL * CCH * 2);
  f16*   vv    = (f16*)  alloc((size_t)NB * CCH * LL * 2);
  float* attn  = (float*)alloc((size_t)NB * CCH * LL * 4);

  k_means<<<dim3(NB * CCH), 64, 0, stream>>>(x, xh_in, xw_in);
  k_wcast<<<dim3(768), 256, 0, stream>>>(w_qkv, w16);
  k_ela<<<dim3(NB * 2), 256, 0, stream>>>(xh_in, xw_in, w1, b1, gam, bet, x_h, x_w);
  k_qkv<<<dim3(32, NB), 256, 0, stream>>>(x, w16, qT, kT, vv);
  k_flash<<<dim3(32, NB), 256, 0, stream>>>(qT, kT, vv, attn);
  k_final<<<dim3((NB * CCH * LL) / 256), 256, 0, stream>>>(x, x_h, x_w, attn, out);
}
